// ContinuousLSTMNextEventModel_54992761258919
// MI455X (gfx1250) — matrix operand order repaired and checked by reading, not compiled, not run on hardware
//
#include <hip/hip_runtime.h>
#include <hip/hip_bf16.h>

// Problem sizes
#define B_  256
#define T_  512
#define V_  10000
#define E_  512
#define H_  1024
#define S_  64
#define EP2 (E_ + 2)
#define HS  (H_ + S_)

typedef __attribute__((ext_vector_type(16))) __bf16 v16bf;
typedef __attribute__((ext_vector_type(8)))  __bf16 v8bf;
typedef __attribute__((ext_vector_type(8)))  float  v8f;

__device__ __forceinline__ v8f wmma_bf16(v16bf a, v16bf b, v8f c) {
  return __builtin_amdgcn_wmma_f32_16x16x32_bf16(false, a, false, b, (short)0, c, false, false);
}

__device__ __forceinline__ unsigned short f2bf(float x) {
  unsigned int u = __float_as_uint(x);
  unsigned int r = (u + 0x7FFFu + ((u >> 16) & 1u)) >> 16;
  return (unsigned short)r;
}
__device__ __forceinline__ float bf2f(unsigned short h) {
  return __uint_as_float(((unsigned int)h) << 16);
}
__device__ __forceinline__ float sigm(float x) { return 1.0f / (1.0f + __expf(-x)); }

// CDNA5 async global->LDS copy (ASYNCcnt-tracked), 16 bytes per lane.
__device__ __forceinline__ void async_cp_b128(void* lds_dst, const void* gsrc) {
  unsigned ldsoff = (unsigned)(size_t)lds_dst;
  unsigned long long ga = (unsigned long long)gsrc;
  asm volatile("global_load_async_to_lds_b128 %0, %1, off"
               :: "v"(ldsoff), "v"(ga) : "memory");
}
__device__ __forceinline__ void wait_async() {
  asm volatile("s_wait_asynccnt 0x0" ::: "memory");
}

// 16x32 bf16 A-fragment from an LDS tile with row stride 32 (contiguous k = 16*half + s).
__device__ __forceinline__ v16bf afrag_lds(const unsigned short* tile, int lane) {
  int m = lane & 15;
  int kb = (lane >> 4) * 16;
  const unsigned short* row = tile + m * 32;
  v8bf lo = *(const v8bf*)(row + kb);
  v8bf hi = *(const v8bf*)(row + kb + 8);
  return __builtin_shufflevector(lo, hi, 0,1,2,3,4,5,6,7,8,9,10,11,12,13,14,15);
}

// 32x16 bf16 B-fragment from a row-major (N x K) weight.
__device__ __forceinline__ v16bf bfrag_w(const unsigned short* W, long long rowstride,
                                         int nbase, int kc, int lane) {
  int n = nbase + (lane & 15);
  int koff = kc + (lane >> 4) * 16;
  return *(const v16bf*)(W + (long long)n * rowstride + koff);
}

// ---------------------------------------------------------------------------
__global__ __launch_bounds__(256) void cvt_bf16_kernel(const float* __restrict__ src,
                                                       unsigned short* __restrict__ dst,
                                                       long long n, int cols, int ld) {
  long long i = (long long)blockIdx.x * blockDim.x + threadIdx.x;
  if (i >= n) return;
  long long r = i / cols;
  int c = (int)(i - r * cols);
  dst[i] = f2bf(src[r * (long long)ld + c]);
}

__global__ __launch_bounds__(256) void init_kernel(const float* __restrict__ deltas,
                                                   float* __restrict__ h0,
                                                   float* __restrict__ cbuf,
                                                   float* __restrict__ dlog) {
  long long i = (long long)blockIdx.x * blockDim.x + threadIdx.x;
  if (i < (long long)B_ * H_) { h0[i] = 0.0f; cbuf[i] = 0.0f; }
  if (i < (long long)B_ * T_) { dlog[i] = log1pf(deltas[i]); }
}

// ---------------------------------------------------------------------------
// Per-step: xp = bf16( emb[lab]@proj_W[:,:E].T + dlog*W[:,E] + v*W[:,E+1] + b )
// Block 128 thr (4 waves); tile 32 rows x 64 cols; grid (B/32, H/64).
// Double-buffered async emb gather overlapped with WMMA.
__global__ __launch_bounds__(128) void xp_step_kernel(const int* __restrict__ labels,
                                                      const float* __restrict__ dlog,
                                                      const int* __restrict__ lengths,
                                                      const unsigned short* __restrict__ embB,
                                                      const unsigned short* __restrict__ projWB,
                                                      const float* __restrict__ proj_W,
                                                      const float* __restrict__ proj_b,
                                                      unsigned short* __restrict__ xp,
                                                      int t) {
  const int b0 = blockIdx.x * 32;
  const int j0 = blockIdx.y * 64;
  const int tid = threadIdx.x;
  const int lane = tid & 31, wid = tid >> 5;
  __shared__ int sLab[32];
  __shared__ unsigned short sA[2][32 * 32];
  if (tid < 32) sLab[tid] = labels[(long long)(b0 + tid) * T_ + t];
  __syncthreads();

  // per-thread async chunk: 1024 elems = 128 x 8-elem chunks
  const int am = tid >> 2;
  const int ak = (tid & 3) * 8;
  const unsigned short* gsrc = &embB[(long long)sLab[am] * E_ + ak];  // hoisted base
  unsigned short* ldst0 = &sA[0][am * 32 + ak];
  unsigned short* ldst1 = &sA[1][am * 32 + ak];

  v8f acc0 = {}, acc1 = {};
  const int nb = j0 + wid * 16;
  constexpr int NIT = E_ / 32;

  async_cp_b128(ldst0, gsrc);                       // prologue: tile 0
  for (int i = 0; i < NIT; ++i) {
    const int cur = i & 1;
    wait_async();
    __syncthreads();
    if (i + 1 < NIT)                                 // stage next tile
      async_cp_b128(cur ? ldst0 : ldst1, gsrc + (i + 1) * 32);
    const unsigned short* tp = sA[cur];
    v16bf a0 = afrag_lds(tp, lane);
    v16bf a1 = afrag_lds(tp + 16 * 32, lane);
    v16bf bw = bfrag_w(projWB, E_, nb, i * 32, lane);
    acc0 = wmma_bf16(a0, bw, acc0);
    acc1 = wmma_bf16(a1, bw, acc1);
    __syncthreads();
  }

  const int n = nb + (lane & 15);
  const int moff = 8 * (lane >> 4);
  const float wE = proj_W[(long long)n * EP2 + E_];
  const float wV = proj_W[(long long)n * EP2 + E_ + 1];
  const float pb = proj_b[n];
#pragma unroll
  for (int rt = 0; rt < 2; ++rt) {
    const v8f& acc = rt ? acc1 : acc0;
#pragma unroll
    for (int r = 0; r < 8; ++r) {
      int b = b0 + rt * 16 + moff + r;
      float dl = dlog[(long long)b * T_ + t];
      float vv = (t < lengths[b]) ? 1.0f : 0.0f;
      xp[(long long)b * H_ + n] = f2bf(acc[r] + dl * wE + vv * wV + pb);
    }
  }
}

// ---------------------------------------------------------------------------
// Per-step fused gates + LSTM cell update (16 WMMAs per 32-wide K chunk).
// Double-buffered: async xp tile + VALU-staged decayed-h tile for chunk i+1
// overlap the WMMAs of chunk i. Grid (B/32, H/64); h double-buffered.
__global__ __launch_bounds__(128) void gates_step_kernel(const unsigned short* __restrict__ xp,
                                                         const float* __restrict__ h_in,
                                                         float* __restrict__ h_out,
                                                         float* __restrict__ cbuf,
                                                         const unsigned short* __restrict__ WihB,
                                                         const unsigned short* __restrict__ WhhB,
                                                         const float* __restrict__ b_ih,
                                                         const float* __restrict__ b_hh,
                                                         const float* __restrict__ dec_W,
                                                         const float* __restrict__ dec_b,
                                                         const float* __restrict__ dlog,
                                                         const int* __restrict__ lengths,
                                                         int t) {
  const int b0 = blockIdx.x * 32;
  const int j0 = blockIdx.y * 64;
  const int tid = threadIdx.x;
  const int lane = tid & 31, wid = tid >> 5;
  __shared__ unsigned short sXp[2][32 * 32];
  __shared__ unsigned short sHd[2][32 * 32];
  __shared__ float sDlog[32];
  if (tid < 32) sDlog[tid] = dlog[(long long)(b0 + tid) * T_ + t];
  __syncthreads();

  const int am = tid >> 2;
  const int ak = (tid & 3) * 8;
  const float dlv = sDlog[am];                               // hoisted to reg
  const unsigned short* xsrc = &xp[(long long)(b0 + am) * H_ + ak];
  const float* hsrc = &h_in[(long long)(b0 + am) * H_ + ak];
  const float* dwB = dec_W + ak;
  const float* dbB = dec_b + ak;
  unsigned short* xd[2] = { &sXp[0][am * 32 + ak], &sXp[1][am * 32 + ak] };
  unsigned short* hd[2] = { &sHd[0][am * 32 + ak], &sHd[1][am * 32 + ak] };

  auto stage = [&](int kc, int buf) {
    async_cp_b128(xd[buf], xsrc + kc);
    const float* hp = hsrc + kc;
    float4 h0 = *(const float4*)hp;
    float4 h1 = *(const float4*)(hp + 4);
    const float* dwp = dwB + kc;
    const float* dbp = dbB + kc;
    ushort4 o0, o1;
    o0.x = f2bf(h0.x * sigm(dlv * dwp[0] + dbp[0]));
    o0.y = f2bf(h0.y * sigm(dlv * dwp[1] + dbp[1]));
    o0.z = f2bf(h0.z * sigm(dlv * dwp[2] + dbp[2]));
    o0.w = f2bf(h0.w * sigm(dlv * dwp[3] + dbp[3]));
    o1.x = f2bf(h1.x * sigm(dlv * dwp[4] + dbp[4]));
    o1.y = f2bf(h1.y * sigm(dlv * dwp[5] + dbp[5]));
    o1.z = f2bf(h1.z * sigm(dlv * dwp[6] + dbp[6]));
    o1.w = f2bf(h1.w * sigm(dlv * dwp[7] + dbp[7]));
    *(ushort4*)hd[buf] = o0;
    *(ushort4*)(hd[buf] + 4) = o1;
  };

  v8f acc[4][2] = {};
  const int nb = j0 + wid * 16;
  constexpr int NIT = H_ / 32;

  stage(0, 0);                                               // prologue
  for (int i = 0; i < NIT; ++i) {
    const int cur = i & 1;
    wait_async();
    __syncthreads();
    if (i + 1 < NIT) stage((i + 1) * 32, cur ^ 1);
    const int kc = i * 32;
    v16bf aX0 = afrag_lds(sXp[cur], lane);
    v16bf aX1 = afrag_lds(sXp[cur] + 16 * 32, lane);
    v16bf aH0 = afrag_lds(sHd[cur], lane);
    v16bf aH1 = afrag_lds(sHd[cur] + 16 * 32, lane);
#pragma unroll
    for (int g = 0; g < 4; ++g) {
      int nB = g * H_ + nb;
      v16bf bi = bfrag_w(WihB, H_, nB, kc, lane);
      v16bf bh = bfrag_w(WhhB, H_, nB, kc, lane);
      acc[g][0] = wmma_bf16(aX0, bi, acc[g][0]);
      acc[g][1] = wmma_bf16(aX1, bi, acc[g][1]);
      acc[g][0] = wmma_bf16(aH0, bh, acc[g][0]);
      acc[g][1] = wmma_bf16(aH1, bh, acc[g][1]);
    }
    __syncthreads();
  }

  const int n = nb + (lane & 15);
  const int moff = 8 * (lane >> 4);
  const float dw = dec_W[n], db = dec_b[n];
  const float bi0 = b_ih[n] + b_hh[n];
  const float bi1 = b_ih[H_ + n] + b_hh[H_ + n];
  const float bi2 = b_ih[2 * H_ + n] + b_hh[2 * H_ + n];
  const float bi3 = b_ih[3 * H_ + n] + b_hh[3 * H_ + n];
#pragma unroll
  for (int rt = 0; rt < 2; ++rt) {
#pragma unroll
    for (int r = 0; r < 8; ++r) {
      int b = b0 + rt * 16 + moff + r;
      float iv = acc[0][rt][r] + bi0;
      float fv = acc[1][rt][r] + bi1;
      float gv = acc[2][rt][r] + bi2;
      float ov = acc[3][rt][r] + bi3;
      float co = cbuf[(long long)b * H_ + n];
      float dl = dlog[(long long)b * T_ + t];
      float hd2 = h_in[(long long)b * H_ + n] * sigm(dl * dw + db);
      float nc = sigm(fv) * co + sigm(iv) * tanhf(gv);
      float nh = sigm(ov) * tanhf(nc);
      bool valid = (t < lengths[b]);
      h_out[(long long)b * H_ + n] = valid ? nh : hd2;
      cbuf[(long long)b * H_ + n] = valid ? nc : co;
    }
  }
}

// ---------------------------------------------------------------------------
__global__ __launch_bounds__(256) void concat_kernel(const float* __restrict__ h,
                                                     const float* __restrict__ state,
                                                     unsigned short* __restrict__ out) {
  long long i = (long long)blockIdx.x * blockDim.x + threadIdx.x;
  if (i >= (long long)B_ * HS) return;
  long long b = i / HS;
  int c = (int)(i - b * HS);
  float v = (c < H_) ? h[b * H_ + c] : state[b * S_ + (c - H_)];
  out[i] = f2bf(v);
}

// Generic bf16 GEMM: out = act(A(MxK) @ W(NxK)^T + bias). M=256 via grid.x.
// Block 128 thr (4 waves); tile 32 rows x 64 cols; grid (M/32, ceil(N/64)).
__global__ __launch_bounds__(128) void gemm_bf16_kernel(const unsigned short* __restrict__ A,
                                                        const unsigned short* __restrict__ W,
                                                        const float* __restrict__ bias,
                                                        float* __restrict__ outF,
                                                        unsigned short* __restrict__ outB,
                                                        int N, int K, int ldo, int relu) {
  const int b0 = blockIdx.x * 32;
  const int j0 = blockIdx.y * 64;
  const int tid = threadIdx.x;
  const int lane = tid & 31, wid = tid >> 5;
  __shared__ unsigned short sA[2][32 * 32];
  const int nb = j0 + wid * 16;
  const bool active = (nb < N);
  const int am = tid >> 2;
  const int ak = (tid & 3) * 8;
  const unsigned short* gsrc = &A[(long long)(b0 + am) * K + ak];
  unsigned short* ldst[2] = { &sA[0][am * 32 + ak], &sA[1][am * 32 + ak] };

  v8f acc0 = {}, acc1 = {};
  const int nIter = K / 32;
  async_cp_b128(ldst[0], gsrc);
  for (int i = 0; i < nIter; ++i) {
    const int cur = i & 1;
    wait_async();
    __syncthreads();
    if (i + 1 < nIter) async_cp_b128(ldst[cur ^ 1], gsrc + (i + 1) * 32);
    if (active) {
      v16bf a0 = afrag_lds(sA[cur], lane);
      v16bf a1 = afrag_lds(sA[cur] + 16 * 32, lane);
      v16bf bw = bfrag_w(W, K, nb, i * 32, lane);
      acc0 = wmma_bf16(a0, bw, acc0);
      acc1 = wmma_bf16(a1, bw, acc1);
    }
    __syncthreads();
  }

  if (active) {
    const int n = nb + (lane & 15);
    const int moff = 8 * (lane >> 4);
    const float bb = bias[n];
#pragma unroll
    for (int rt = 0; rt < 2; ++rt) {
      const v8f& acc = rt ? acc1 : acc0;
#pragma unroll
      for (int r = 0; r < 8; ++r) {
        int b = b0 + rt * 16 + moff + r;
        float v = acc[r] + bb;
        if (relu) v = fmaxf(v, 0.0f);
        if (outF) outF[(long long)b * ldo + n] = v;
        if (outB) outB[(long long)b * N + n] = f2bf(v);
      }
    }
  }
}

// delta head: out[b] = dot(t2[b], del_W) + del_b
__global__ __launch_bounds__(256) void del_kernel(const unsigned short* __restrict__ t2,
                                                  const float* __restrict__ del_W,
                                                  const float* __restrict__ del_b,
                                                  float* __restrict__ out) {
  int b = threadIdx.x;
  if (b >= B_) return;
  float s = 0.0f;
  for (int k = 0; k < H_; ++k) s += bf2f(t2[(long long)b * H_ + k]) * del_W[k];
  out[b] = s + del_b[0];
}

// ---------------------------------------------------------------------------
extern "C" void kernel_launch(void* const* d_in, const int* in_sizes, int n_in,
                              void* d_out, int out_size, void* d_ws, size_t ws_size,
                              hipStream_t stream) {
  const int*   labels  = (const int*)d_in[0];
  const float* deltas  = (const float*)d_in[1];
  const int*   lengths = (const int*)d_in[2];
  const float* state   = (const float*)d_in[3];
  const float* emb     = (const float*)d_in[4];
  const float* proj_W  = (const float*)d_in[5];
  const float* proj_b  = (const float*)d_in[6];
  const float* dec_W   = (const float*)d_in[7];
  const float* dec_b   = (const float*)d_in[8];
  const float* W_ih    = (const float*)d_in[9];
  const float* W_hh    = (const float*)d_in[10];
  const float* b_ih    = (const float*)d_in[11];
  const float* b_hh    = (const float*)d_in[12];
  const float* sh1_W   = (const float*)d_in[13];
  const float* sh1_b   = (const float*)d_in[14];
  const float* sh2_W   = (const float*)d_in[15];
  const float* sh2_b   = (const float*)d_in[16];
  const float* lab_W   = (const float*)d_in[17];
  const float* lab_b   = (const float*)d_in[18];
  const float* del_W   = (const float*)d_in[19];
  const float* del_b   = (const float*)d_in[20];
  float* out = (float*)d_out;

  char* base = (char*)d_ws;
  size_t off = 0;
  auto alloc = [&](size_t bytes) -> void* {
    void* r = base + off;
    off = (off + bytes + 255) & ~(size_t)255;
    return r;
  };
  unsigned short* embB    = (unsigned short*)alloc((size_t)V_ * E_ * 2);
  unsigned short* projWB  = (unsigned short*)alloc((size_t)H_ * E_ * 2);
  unsigned short* WihB    = (unsigned short*)alloc((size_t)4 * H_ * H_ * 2);
  unsigned short* WhhB    = (unsigned short*)alloc((size_t)4 * H_ * H_ * 2);
  unsigned short* sh1WB   = (unsigned short*)alloc((size_t)H_ * HS * 2);
  unsigned short* sh2WB   = (unsigned short*)alloc((size_t)H_ * H_ * 2);
  unsigned short* labWB   = (unsigned short*)alloc((size_t)V_ * H_ * 2);
  unsigned short* xpB     = (unsigned short*)alloc((size_t)B_ * H_ * 2);
  unsigned short* concatB = (unsigned short*)alloc((size_t)B_ * HS * 2);
  unsigned short* t1B     = (unsigned short*)alloc((size_t)B_ * H_ * 2);
  unsigned short* t2B     = (unsigned short*)alloc((size_t)B_ * H_ * 2);
  float* hA   = (float*)alloc((size_t)B_ * H_ * 4);
  float* hB   = (float*)alloc((size_t)B_ * H_ * 4);
  float* cbuf = (float*)alloc((size_t)B_ * H_ * 4);
  float* dlog = (float*)alloc((size_t)B_ * T_ * 4);

  auto cvt = [&](const float* src, unsigned short* dst, long long rows, long long cols, long long ld) {
    long long n = rows * cols;
    cvt_bf16_kernel<<<dim3((unsigned)((n + 255) / 256)), dim3(256), 0, stream>>>(src, dst, n, (int)cols, (int)ld);
  };
  cvt(emb,    embB,   V_,    E_,  E_);
  cvt(proj_W, projWB, H_,    E_,  EP2);   // strip the 2 extra columns
  cvt(W_ih,   WihB,   4*H_,  H_,  H_);
  cvt(W_hh,   WhhB,   4*H_,  H_,  H_);
  cvt(sh1_W,  sh1WB,  H_,    HS,  HS);
  cvt(sh2_W,  sh2WB,  H_,    H_,  H_);
  cvt(lab_W,  labWB,  V_,    H_,  H_);

  {
    long long n = (long long)B_ * H_;  // >= B_*T_
    init_kernel<<<dim3((unsigned)((n + 255) / 256)), dim3(256), 0, stream>>>(deltas, hA, cbuf, dlog);
  }

  const dim3 tileGrid(B_ / 32, H_ / 64);
  for (int t = 0; t < T_; ++t) {
    xp_step_kernel<<<tileGrid, 128, 0, stream>>>(labels, dlog, lengths, embB, projWB,
                                                 proj_W, proj_b, xpB, t);
    const float* hin  = (t & 1) ? hB : hA;
    float*       hout = (t & 1) ? hA : hB;
    gates_step_kernel<<<tileGrid, 128, 0, stream>>>(xpB, hin, hout, cbuf, WihB, WhhB,
                                                    b_ih, b_hh, dec_W, dec_b, dlog, lengths, t);
  }
  // T_ = 512 even => final h lives in hA (t=511 writes hA)

  {
    long long n = (long long)B_ * HS;
    concat_kernel<<<dim3((unsigned)((n + 255) / 256)), dim3(256), 0, stream>>>(hA, state, concatB);
  }
  gemm_bf16_kernel<<<dim3(B_ / 32, H_ / 64), 128, 0, stream>>>(concatB, sh1WB, sh1_b,
                                                               nullptr, t1B, H_, HS, 0, 1);
  gemm_bf16_kernel<<<dim3(B_ / 32, H_ / 64), 128, 0, stream>>>(t1B, sh2WB, sh2_b,
                                                               nullptr, t2B, H_, H_, 0, 1);
  gemm_bf16_kernel<<<dim3(B_ / 32, (V_ + 63) / 64), 128, 0, stream>>>(t2B, labWB, lab_b,
                                                                      out, nullptr, V_, H_, V_, 0);
  del_kernel<<<1, 256, 0, stream>>>(t2B, del_W, del_b, out + (size_t)B_ * V_);

  (void)in_sizes; (void)n_in; (void)out_size; (void)ws_size;
}
